// CosineEdgeExtractor_9663676416634
// MI455X (gfx1250) — matrix operand order repaired and checked by reading, not compiled, not run on hardware
//
#include <hip/hip_runtime.h>
#include <hip/hip_bf16.h>

// CDNA5 / gfx1250: wave32, WMMA f32_16x16x32_f16 path.
typedef __attribute__((ext_vector_type(16))) _Float16 v16h_t;
typedef __attribute__((ext_vector_type(8)))  float    v8f_t;

#define B_DIM 256
#define A_DIM 64
#define S_DIM 512
#define L_DIM 2048
#define K_SEL 16

// A-matrix 16x32 f16 fragment (ISA 7.12.2): lane holds row M=lane&15,
// K chunks at kb..kb+7 and kb+16..kb+23 where kb = (lane>>4)*8.
// p already points at row_base + k0 + (lane>>4)*8.
__device__ __forceinline__ v16h_t load_frag_split(const float* __restrict__ p,
                                                  float& ssq) {
  const float4* q0 = (const float4*)(p);
  const float4* q1 = (const float4*)(p + 16);
  float4 c0 = q0[0], c1 = q0[1];
  float4 c2 = q1[0], c3 = q1[1];
  float f[16] = {c0.x, c0.y, c0.z, c0.w, c1.x, c1.y, c1.z, c1.w,
                 c2.x, c2.y, c2.z, c2.w, c3.x, c3.y, c3.z, c3.w};
  v16h_t r;
#pragma unroll
  for (int i = 0; i < 16; ++i) {
    ssq = fmaf(f[i], f[i], ssq);
    r[i] = (_Float16)f[i];
  }
  return r;
}

// B-matrix 32x16 f16 fragment: lane holds col N=lane&15, 16 contiguous K
// values starting at (lane>>4)*16 (mirrors sparse-B layout in ISA 7.12.4).
// p already points at row_base + k0 + (lane>>4)*16.
__device__ __forceinline__ v16h_t load_frag_contig(const float* __restrict__ p,
                                                   float& ssq) {
  const float4* q = (const float4*)(p);
  float4 c0 = q[0], c1 = q[1], c2 = q[2], c3 = q[3];
  float f[16] = {c0.x, c0.y, c0.z, c0.w, c1.x, c1.y, c1.z, c1.w,
                 c2.x, c2.y, c2.z, c2.w, c3.x, c3.y, c3.z, c3.w};
  v16h_t r;
#pragma unroll
  for (int i = 0; i < 16; ++i) {
    ssq = fmaf(f[i], f[i], ssq);
    r[i] = (_Float16)f[i];
  }
  return r;
}

__global__ __launch_bounds__(256) void cosine_edge_topk_kernel(
    const float* __restrict__ Xa,   // (B, A, L) f32
    const float* __restrict__ Xs,   // (B, S, L) f32
    float* __restrict__ out) {      // edges (B,2,A*k) then weights (B,A*k)
  __shared__ float s_dots[16 * S_DIM];   // 32 KB raw dot tile
  __shared__ float s_invns[S_DIM];
  __shared__ float s_invna[16];

  const int wg   = blockIdx.x;
  const int b    = wg >> 2;        // batch
  const int rb   = wg & 3;         // 16-row block of actuators
  const int tid  = threadIdx.x;
  const int wave = tid >> 5;       // 8 waves
  const int lane = tid & 31;
  const int l15  = lane & 15;
  const int hi   = lane >> 4;

  // A fragment base: row = rb*16 + (lane&15), K offset (lane>>4)*8
  const float* aBase =
      Xa + ((size_t)b * A_DIM + rb * 16 + l15) * L_DIM + hi * 8;

  // Each wave owns 64 sensors: 4 tiles of 16.
  const int n0 = wave * 64;
  const float* sBase[4];
#pragma unroll
  for (int t = 0; t < 4; ++t)
    sBase[t] = Xs + ((size_t)b * S_DIM + n0 + t * 16 + l15) * L_DIM + hi * 8;

  v8f_t acc[4] = {v8f_t{}, v8f_t{}, v8f_t{}, v8f_t{}};
  float ssqA = 0.f;
  float ssqS[4] = {0.f, 0.f, 0.f, 0.f};

  // K loop: 2048 / 32 = 64 steps, 4 WMMAs per step per wave.
  for (int k0 = 0; k0 < L_DIM; k0 += 32) {
    v16h_t af = load_frag_split(aBase + k0, ssqA);
#pragma unroll
    for (int t = 0; t < 4; ++t) {
      v16h_t bf = load_frag_split(sBase[t] + k0, ssqS[t]);
      acc[t] = __builtin_amdgcn_wmma_f32_16x16x32_f16(
          false, af, false, bf, (short)0, acc[t], false, false);
    }
  }

  // Row-norm of actuators: lane L and L^16 together cover all K.
  {
    float tot = ssqA + __shfl_xor(ssqA, 16);
    if (wave == 0 && lane < 16) s_invna[lane] = rsqrtf(tot);
  }

  const int mBase = hi * 8;  // C layout: VGPR i -> M = i + 8*(lane>>4)
#pragma unroll
  for (int t = 0; t < 4; ++t) {
    float tot = ssqS[t] + __shfl_xor(ssqS[t], 16);
    if (lane < 16) s_invns[n0 + t * 16 + lane] = rsqrtf(tot);
    const int n = n0 + t * 16 + l15;  // C layout: N = lane&15
#pragma unroll
    for (int i = 0; i < 8; ++i)
      s_dots[(mBase + i) * S_DIM + n] = acc[t][i];
  }
  __syncthreads();

  // ---- Top-k (iterative selection, descending, lowest-index tie-break) ----
  for (int row = wave; row < 16; row += 8) {
    const float inva = s_invna[row];
    float simv[16], simsq[16];
#pragma unroll
    for (int j = 0; j < 16; ++j) {
      const int n = lane + 32 * j;
      const float v = s_dots[row * S_DIM + n] * inva * s_invns[n];
      simv[j]  = v;
      simsq[j] = v * v;
    }
    const int mg = rb * 16 + row;   // global actuator index 0..63
    for (int sel = 0; sel < K_SEL; ++sel) {
      float bv = -1.f, bval = 0.f;
      int bidx = 0;
#pragma unroll
      for (int j = 0; j < 16; ++j) {
        if (simsq[j] > bv) {
          bv = simsq[j];
          bidx = lane + 32 * j;
          bval = simv[j];
        }
      }
#pragma unroll
      for (int off = 16; off > 0; off >>= 1) {
        const float ov  = __shfl_xor(bv, off);
        const int   oi  = __shfl_xor(bidx, off);
        const float obv = __shfl_xor(bval, off);
        if (ov > bv || (ov == bv && oi < bidx)) {
          bv = ov; bidx = oi; bval = obv;
        }
      }
#pragma unroll
      for (int j = 0; j < 16; ++j)
        if (bidx == lane + 32 * j) simsq[j] = -1.f;  // knock out winner
      if (lane == 0) {
        const int e = mg * K_SEL + sel;
        const size_t eb = (size_t)b * (2 * A_DIM * K_SEL);
        out[eb + e] = (float)mg;                         // edges[b][0][e]
        out[eb + A_DIM * K_SEL + e] = (float)bidx;       // edges[b][1][e]
        out[(size_t)B_DIM * 2 * A_DIM * K_SEL +
            (size_t)b * (A_DIM * K_SEL) + e] = bval;     // weights[b][e]
      }
    }
  }
}

extern "C" void kernel_launch(void* const* d_in, const int* in_sizes, int n_in,
                              void* d_out, int out_size, void* d_ws,
                              size_t ws_size, hipStream_t stream) {
  (void)in_sizes; (void)n_in; (void)d_ws; (void)ws_size; (void)out_size;
  const float* Xa = (const float*)d_in[0];
  const float* Xs = (const float*)d_in[1];
  float* out = (float*)d_out;
  dim3 grid(B_DIM * (A_DIM / 16));   // 1024 workgroups
  dim3 block(256);                   // 8 wave32
  hipLaunchKernelGGL(cosine_edge_topk_kernel, grid, block, 0, stream,
                     Xa, Xs, out);
}
